// Model_26353919328765
// MI455X (gfx1250) — matrix operand order repaired and checked by reading, not compiled, not run on hardware
//
#include <hip/hip_runtime.h>

typedef __bf16 bf16_t;
typedef __attribute__((ext_vector_type(16))) bf16_t v16bf;
typedef __attribute__((ext_vector_type(8)))  float  v8f;
typedef unsigned int v4u __attribute__((ext_vector_type(4)));
typedef int v8i __attribute__((ext_vector_type(8)));
typedef int v4i __attribute__((ext_vector_type(4)));

#define WMMA_BF16(A,B,C) __builtin_amdgcn_wmma_f32_16x16x32_bf16(false,(A),false,(B),(short)0,(C),false,false)

#define BB 64
#define TT 512
#define JJ 64
#define DD 256

// ---------- fragment loaders (per CDNA5 ISA 7.12.2 layouts) ----------
// A (16xK, 16-bit): lane L -> row L&15; VGPR0-3 = K (L>>4)*8 .. +7; VGPR4-7 = +16..+23
__device__ __forceinline__ v16bf frag_a(const bf16_t* p, int ld) {
  int lane = threadIdx.x & 31;
  const bf16_t* q = p + (size_t)(lane & 15) * ld + ((lane >> 4) << 3);
  union { v16bf v; uint4 u[2]; } r;
  r.u[0] = *(const uint4*)(q);
  r.u[1] = *(const uint4*)(q + 16);
  return r.v;
}
// B (Kx16, 16-bit) from N-major (NxK row-major) memory: lane L -> col L&15,
// VGPR0-3 = K (L>>4)*8 .. +7; VGPR4-7 = +16..+23
__device__ __forceinline__ v16bf frag_b(const bf16_t* p, int ld) {
  int lane = threadIdx.x & 31;
  const bf16_t* q = p + (size_t)(lane & 15) * ld + ((lane >> 4) << 3);
  union { v16bf v; uint4 u[2]; } r;
  r.u[0] = *(const uint4*)(q);
  r.u[1] = *(const uint4*)(q + 16);
  return r.v;
}

__device__ __forceinline__ float sigf(float x){ return 1.0f/(1.0f+__expf(-x)); }

// ---------- small prep kernels ----------
__global__ void k_f2bf(const float* __restrict__ s, bf16_t* __restrict__ d, int n){
  int i = blockIdx.x*blockDim.x + threadIdx.x;
  if (i < n) d[i] = (bf16_t)s[i];
}
__global__ void k_addvec(const float* __restrict__ a, const float* __restrict__ b,
                         float* __restrict__ o, int n){
  int i = blockIdx.x*blockDim.x + threadIdx.x;
  if (i < n) o[i] = a[i] + b[i];
}
__global__ void k_prep_q(const float* __restrict__ q, bf16_t* __restrict__ qbf,
                         bf16_t* __restrict__ qTbf){
  int i = blockIdx.x*blockDim.x + threadIdx.x;   // < B*J*D
  bf16_t h = (bf16_t)q[i];
  qbf[i] = h;
  int d = i & 255, j = (i >> 8) & 63, b = i >> 14;
  qTbf[((size_t)b*DD + d)*JJ + j] = h;
}
__global__ void k_prep_c(const float* __restrict__ c, const float* __restrict__ wcq,
                         bf16_t* __restrict__ o){
  int i = blockIdx.x*blockDim.x + threadIdx.x;   // < B*T*D
  o[i] = (bf16_t)(c[i] * wcq[i & 255]);
}
__global__ void k_rowdot(const float* __restrict__ X, const float* __restrict__ w,
                         const float* bias1, const float* bias2,
                         float* __restrict__ o, int rows){
  int r = blockIdx.x*blockDim.x + threadIdx.x;
  if (r >= rows) return;
  const float* x = X + (size_t)r*DD;
  float acc = bias1[0] + (bias2 ? bias2[0] : 0.0f);
  #pragma unroll 8
  for (int d = 0; d < DD; d++) acc += x[d]*w[d];
  o[r] = acc;
}

// ---------- attention: scores + softmax(J) + c2q, one wave per 16 rows ----------
__global__ void __launch_bounds__(128) k_attn(
    const bf16_t* __restrict__ cwbf, const bf16_t* __restrict__ qbf,
    const bf16_t* __restrict__ qTbf, const float* __restrict__ cwv,
    const float* __restrict__ qwv, float* __restrict__ smax, float* __restrict__ c2q){
  __shared__ __align__(16) bf16_t aLds[4][16*64];
  int wv  = threadIdx.x >> 5;
  int wid = blockIdx.x*4 + wv;
  int b   = wid >> 5;            // 32 t-tiles per batch
  int t0  = (wid & 31) << 4;
  int lane = threadIdx.x & 31;
  int colL = lane & 15;
  int rh   = (lane >> 4) << 3;

  const bf16_t* Ab = cwbf + ((size_t)b*TT + t0)*DD;
  const bf16_t* Qb = qbf + (size_t)b*JJ*DD;
  v8f s[4];
  #pragma unroll
  for (int i=0;i<4;i++) for (int r=0;r<8;r++) s[i][r] = 0.f;
  for (int k0=0;k0<DD;k0+=32){
    v16bf a = frag_a(Ab + k0, DD);
    #pragma unroll
    for (int jt=0;jt<4;jt++){
      v16bf bb = frag_b(Qb + (size_t)(jt*16)*DD + k0, DD);
      s[jt] = WMMA_BF16(a, bb, s[jt]);
    }
  }
  float rmax[8], cwl[8];
  #pragma unroll
  for (int r=0;r<8;r++){ cwl[r] = cwv[(size_t)b*TT + t0 + r + rh]; rmax[r] = -1e30f; }
  #pragma unroll
  for (int jt=0;jt<4;jt++){
    float qv = qwv[b*JJ + jt*16 + colL];
    for (int r=0;r<8;r++){
      float v = s[jt][r] + cwl[r] + qv;
      s[jt][r] = v;
      rmax[r] = fmaxf(rmax[r], v);
    }
  }
  // reduce across the 16 column lanes (xor < 16 stays within each half)
  #pragma unroll
  for (int m=8;m;m>>=1) for (int r=0;r<8;r++) rmax[r] = fmaxf(rmax[r], __shfl_xor(rmax[r], m, 32));
  if (colL == 0)
    for (int r=0;r<8;r++) smax[(size_t)b*TT + t0 + r + rh] = rmax[r];
  float rsum[8];
  #pragma unroll
  for (int r=0;r<8;r++) rsum[r] = 0.f;
  #pragma unroll
  for (int jt=0;jt<4;jt++) for (int r=0;r<8;r++){
    float e = __expf(s[jt][r] - rmax[r]); s[jt][r] = e; rsum[r] += e;
  }
  #pragma unroll
  for (int m=8;m;m>>=1) for (int r=0;r<8;r++) rsum[r] += __shfl_xor(rsum[r], m, 32);
  #pragma unroll
  for (int r=0;r<8;r++) rsum[r] = 1.0f / rsum[r];
  bf16_t* al = aLds[wv];
  #pragma unroll
  for (int jt=0;jt<4;jt++) for (int r=0;r<8;r++)
    al[(r+rh)*64 + jt*16 + colL] = (bf16_t)(s[jt][r]*rsum[r]);
  asm volatile("s_wait_dscnt 0x0" ::: "memory");   // wave-local LDS RAW fence
  v8f acc[16];
  #pragma unroll
  for (int i=0;i<16;i++) for (int r=0;r<8;r++) acc[i][r] = 0.f;
  const bf16_t* QT = qTbf + (size_t)b*DD*JJ;       // [D, J] = NxK for B operand
  for (int kt=0;kt<2;kt++){
    v16bf a = frag_a(al + kt*32, 64);
    #pragma unroll
    for (int nt=0;nt<16;nt++){
      v16bf bb = frag_b(QT + (size_t)(nt*16)*JJ + kt*32, JJ);
      acc[nt] = WMMA_BF16(a, bb, acc[nt]);
    }
  }
  #pragma unroll
  for (int nt=0;nt<16;nt++) for (int r=0;r<8;r++)
    c2q[((size_t)b*TT + t0 + r + rh)*DD + nt*16 + colL] = acc[nt][r];
}

// ---------- q2c: softmax over T of row-max, then weighted sum of c ----------
__global__ void __launch_bounds__(256) k_q2c(const float* __restrict__ smax,
                                             const float* __restrict__ c,
                                             float* __restrict__ q2c){
  int b = blockIdx.x, tid = threadIdx.x;
  __shared__ float red[256];
  __shared__ float batt[512];
  float v0 = smax[(size_t)b*TT + tid];
  float v1 = smax[(size_t)b*TT + 256 + tid];
  red[tid] = fmaxf(v0, v1); __syncthreads();
  for (int s=128;s;s>>=1){ if (tid<s) red[tid]=fmaxf(red[tid],red[tid+s]); __syncthreads(); }
  float mx = red[0]; __syncthreads();
  float e0 = __expf(v0-mx), e1 = __expf(v1-mx);
  red[tid] = e0+e1; __syncthreads();
  for (int s=128;s;s>>=1){ if (tid<s) red[tid]+=red[tid+s]; __syncthreads(); }
  float inv = 1.0f/red[0];
  batt[tid] = e0*inv; batt[tid+256] = e1*inv;
  __syncthreads();
  float acc = 0.f;
  for (int t=0;t<TT;t++) acc += batt[t]*c[((size_t)b*TT+t)*DD + tid];
  q2c[b*DD + tid] = acc;
}

// ---------- assemble g = [c, c2q, c*c2q, c*q2c] in bf16 ----------
__global__ void k_g(const float* __restrict__ c, const float* __restrict__ c2q,
                    const float* __restrict__ q2c, bf16_t* __restrict__ gbf){
  int i = blockIdx.x*blockDim.x + threadIdx.x; // < B*T*1024
  int k = i & 1023, row = i >> 10, d = k & 255, seg = k >> 8;
  float cv = c[(size_t)row*DD + d];
  float v;
  if      (seg == 0) v = cv;
  else if (seg == 1) v = c2q[(size_t)row*DD + d];
  else if (seg == 2) v = cv * c2q[(size_t)row*DD + d];
  else               v = cv * q2c[(row >> 9)*DD + d];
  gbf[i] = (bf16_t)v;
}

// ---------- big GEMM: C[M,*] = A[M,K] @ W[512,K]^T + bias, bf16 out ----------
__global__ void __launch_bounds__(256) k_gemm(const bf16_t* __restrict__ A, int K,
                                              const bf16_t* __restrict__ W,
                                              const float* __restrict__ bias,
                                              bf16_t* __restrict__ C, int colofs){
  int wv = threadIdx.x >> 5;
  int lane = threadIdx.x & 31;
  int colL = lane & 15, rh = (lane>>4) << 3;
  int row0 = blockIdx.x*64 + (wv>>1)*16;
  int n0   = blockIdx.y*128 + (wv&1)*64;
  v8f acc[4];
  #pragma unroll
  for (int i=0;i<4;i++) for (int r=0;r<8;r++) acc[i][r]=0.f;
  const bf16_t* Ar = A + (size_t)row0*K;
  for (int k0=0;k0<K;k0+=32){
    v16bf a = frag_a(Ar + k0, K);
    #pragma unroll
    for (int nt=0;nt<4;nt++){
      v16bf bb = frag_b(W + (size_t)(n0+nt*16)*K + k0, K);
      acc[nt] = WMMA_BF16(a, bb, acc[nt]);
    }
  }
  #pragma unroll
  for (int nt=0;nt<4;nt++){
    int col = n0 + nt*16 + colL;
    float bv = bias[col];
    for (int r=0;r<8;r++)
      C[(size_t)(row0 + r + rh)*1024 + colofs + col] = (bf16_t)(acc[nt][r] + bv);
  }
}

// ---------- persistent LSTM recurrence: 1 block = (dir, 16 batches) ----------
// LDS: Whh (128KB, staged by the Tensor Data Mover) + h(bf16) + c(f32) + gates(f32)
__global__ void __launch_bounds__(256) k_lstm(const bf16_t* __restrict__ pre,
                                              const bf16_t* __restrict__ whhF,
                                              const bf16_t* __restrict__ whhB,
                                              bf16_t* __restrict__ hout){
  extern __shared__ char smem[];
  bf16_t* whh   = (bf16_t*)smem;                          // 512x128 bf16 = 128KB @ LDS 0
  bf16_t* hbf   = (bf16_t*)(smem + 131072);               // 16x128 bf16
  float*  cst   = (float*)(smem + 131072 + 4096);         // 16x128 f32
  float*  gates = (float*)(smem + 131072 + 4096 + 8192);  // 16x512 f32

  int dir = blockIdx.x & 1;
  int b0  = (blockIdx.x >> 1) * 16;
  int tid = threadIdx.x;
  const bf16_t* Wsrc = dir ? whhB : whhF;

#if __has_builtin(__builtin_amdgcn_tensor_load_to_lds)
  // TDM: 1-D tensor copy of 131072B (32768 dwords) global -> LDS offset 0.
  if (tid < 32) {                       // one wave issues the DMA descriptor
    unsigned long long ga = (unsigned long long)(const void*)Wsrc;
    v4u g0;
    g0[0] = 1u;                                         // count=1, user mode
    g0[1] = 0u;                                         // lds_addr = 0
    g0[2] = (unsigned)(ga & 0xffffffffu);               // global_addr[31:0]
    g0[3] = (unsigned)((ga >> 32) & 0x01ffffffu) | (2u << 30);  // addr[56:32], type=2
    v8i g1;
    g1[0] = 0x00020000;                                 // data_size=2 (4B), mask=0
    g1[1] = (int)0x80000000;                            // tensor_dim0 = 0x8000 dwords
    g1[2] = 0x00010000;                                 // tensor_dim1 = 1
    g1[3] = (int)0x80000000;                            // tile_dim0 = 0x8000
    g1[4] = 0;                                          // tile_dim1/2 unused
    g1[5] = 0x8000;                                     // tensor_dim0_stride
    g1[6] = 0; g1[7] = 0;
    v4i zz4 = {0,0,0,0};
    v8i zz8 = {0,0,0,0,0,0,0,0};
    __builtin_amdgcn_tensor_load_to_lds(g0, g1, zz4, zz4, zz8, 0);
    __builtin_amdgcn_s_wait_tensorcnt(0);
  }
#else
  {
    const uint4* src4 = (const uint4*)Wsrc;
    uint4* dst4 = (uint4*)whh;
    for (int i = tid; i < 512*128*2/16; i += 256) dst4[i] = src4[i];
  }
#endif
  for (int i = tid; i < 16*128; i += 256){ hbf[i] = (bf16_t)0.0f; cst[i] = 0.f; }
  __syncthreads();

  int wv = tid >> 5, lane = tid & 31;
  int colL = lane & 15, rh = (lane>>4)<<3;
  int n0 = wv * 64;                       // wave's 64 gate columns of 512
  for (int t = 0; t < TT; t++){
    int idx = dir ? (TT - 1 - t) : t;
    // prefetch next step's pre rows while this step's WMMAs run
    if (t + 1 < TT){
      int nidx = dir ? (TT - 2 - t) : (t + 1);
      const bf16_t* np = &pre[(size_t)((b0 + (tid & 15))*TT + nidx)*1024 + dir*512 + ((tid >> 4)*64)];
      __builtin_prefetch(np, 0, 0);
    }
    v8f acc[4];
    #pragma unroll
    for (int nt=0;nt<4;nt++)
      for (int r=0;r<8;r++) acc[nt][r] = 0.f;
    #pragma unroll
    for (int k0=0;k0<128;k0+=32){
      v16bf a = frag_a(hbf + k0, 128);
      #pragma unroll
      for (int nt=0;nt<4;nt++){
        v16bf bb = frag_b(whh + (size_t)(n0+nt*16)*128 + k0, 128);
        acc[nt] = WMMA_BF16(a, bb, acc[nt]);
      }
    }
    #pragma unroll
    for (int nt=0;nt<4;nt++)
      for (int r=0;r<8;r++)
        gates[(r+rh)*512 + n0 + nt*16 + colL] = acc[nt][r];
    __syncthreads();
    #pragma unroll
    for (int e = tid; e < 2048; e += 256){
      int rr = e >> 7, cc = e & 127;
      // coalesced bf16 reads of the (bias-fused) input projection
      size_t pb = (size_t)((b0+rr)*TT + idx)*1024 + dir*512 + cc;
      float gi = gates[rr*512 + cc]       + (float)pre[pb];
      float gf = gates[rr*512 + 128 + cc] + (float)pre[pb + 128];
      float gg = gates[rr*512 + 256 + cc] + (float)pre[pb + 256];
      float go = gates[rr*512 + 384 + cc] + (float)pre[pb + 384];
      float cn = sigf(gf)*cst[rr*128+cc] + sigf(gi)*tanhf(gg);
      float hn = sigf(go)*tanhf(cn);
      cst[rr*128+cc] = cn;
      hbf[rr*128+cc] = (bf16_t)hn;
      hout[(size_t)((b0+rr)*TT + idx)*256 + dir*128 + cc] = (bf16_t)hn;
    }
    __syncthreads();
  }
}

// ---------- pointer logits: z = [g,m] @ W + b ----------
__global__ void k_dot(const bf16_t* __restrict__ gbf, const bf16_t* __restrict__ mbf,
                      const bf16_t* __restrict__ m2bf,
                      const float* __restrict__ W0, const float* __restrict__ b0,
                      const float* __restrict__ W1, const float* __restrict__ b1,
                      float* __restrict__ z1, float* __restrict__ z2){
  int r = blockIdx.x*blockDim.x + threadIdx.x; // < B*T
  union U8 { uint4 u; bf16_t h[8]; };
  const uint4* gp = (const uint4*)(gbf + (size_t)r*1024);
  float s1 = 0.f, s2 = 0.f;
  for (int i=0;i<128;i++){
    U8 w; w.u = gp[i];
    int k = i*8;
    #pragma unroll
    for (int j=0;j<8;j++){ float f = (float)w.h[j]; s1 += f*W0[k+j]; s2 += f*W1[k+j]; }
  }
  const uint4* mp  = (const uint4*)(mbf  + (size_t)r*256);
  const uint4* m2p = (const uint4*)(m2bf + (size_t)r*256);
  for (int i=0;i<32;i++){
    U8 w; w.u = mp[i];
    U8 w2; w2.u = m2p[i];
    int k = 1024 + i*8;
    #pragma unroll
    for (int j=0;j<8;j++){ s1 += (float)w.h[j]*W0[k+j]; s2 += (float)w2.h[j]*W1[k+j]; }
  }
  z1[r] = s1 + b0[0];
  z2[r] = s2 + b1[0];
}

// ---------- softmax over T -> d_out ----------
__global__ void __launch_bounds__(256) k_smaxT(const float* __restrict__ z1,
                                               const float* __restrict__ z2,
                                               float* __restrict__ out){
  int sel = blockIdx.x >> 6;
  int b = blockIdx.x & 63;
  const float* z = sel ? z2 : z1;
  float* o = out + (size_t)sel*(BB*TT) + (size_t)b*TT;
  int tid = threadIdx.x;
  __shared__ float red[256];
  float v0 = z[(size_t)b*TT + tid], v1 = z[(size_t)b*TT + 256 + tid];
  red[tid] = fmaxf(v0, v1); __syncthreads();
  for (int s=128;s;s>>=1){ if (tid<s) red[tid]=fmaxf(red[tid],red[tid+s]); __syncthreads(); }
  float mx = red[0]; __syncthreads();
  float e0=__expf(v0-mx), e1=__expf(v1-mx);
  red[tid]=e0+e1; __syncthreads();
  for (int s=128;s;s>>=1){ if (tid<s) red[tid]+=red[tid+s]; __syncthreads(); }
  float inv = 1.0f/red[0];
  o[tid] = e0*inv; o[tid+256] = e1*inv;
}

extern "C" void kernel_launch(void* const* d_in, const int* in_sizes, int n_in,
                              void* d_out, int out_size, void* d_ws, size_t ws_size,
                              hipStream_t stream) {
  (void)in_sizes; (void)n_in; (void)out_size; (void)ws_size;
  const float* c   = (const float*)d_in[0];
  const float* q   = (const float*)d_in[1];
  const float* wc  = (const float*)d_in[2];
  const float* bc  = (const float*)d_in[3];
  const float* wq  = (const float*)d_in[4];
  const float* bq  = (const float*)d_in[5];
  const float* wcq = (const float*)d_in[6];
  const float* bcq = (const float*)d_in[7];
  const float* l1f_Wih = (const float*)d_in[8];
  const float* l1f_Whh = (const float*)d_in[9];
  const float* l1f_bih = (const float*)d_in[10];
  const float* l1f_bhh = (const float*)d_in[11];
  const float* l1b_Wih = (const float*)d_in[12];
  const float* l1b_Whh = (const float*)d_in[13];
  const float* l1b_bih = (const float*)d_in[14];
  const float* l1b_bhh = (const float*)d_in[15];
  const float* l2f_Wih = (const float*)d_in[16];
  const float* l2f_Whh = (const float*)d_in[17];
  const float* l2f_bih = (const float*)d_in[18];
  const float* l2f_bhh = (const float*)d_in[19];
  const float* l2b_Wih = (const float*)d_in[20];
  const float* l2b_Whh = (const float*)d_in[21];
  const float* l2b_bih = (const float*)d_in[22];
  const float* l2b_bhh = (const float*)d_in[23];
  const float* W0 = (const float*)d_in[24];
  const float* b0 = (const float*)d_in[25];
  const float* W1 = (const float*)d_in[26];
  const float* b1 = (const float*)d_in[27];

  char* ws = (char*)d_ws; size_t off = 0;
  auto alloc = [&](size_t n){ void* p = ws + off; off += (n + 255) & ~(size_t)255; return p; };

  bf16_t* qbf   = (bf16_t*)alloc((size_t)BB*JJ*DD*2);
  bf16_t* qTbf  = (bf16_t*)alloc((size_t)BB*DD*JJ*2);
  bf16_t* cwbf  = (bf16_t*)alloc((size_t)BB*TT*DD*2);
  float*  cw    = (float*) alloc((size_t)BB*TT*4);
  float*  qw    = (float*) alloc((size_t)BB*JJ*4);
  float*  smax  = (float*) alloc((size_t)BB*TT*4);
  float*  c2q   = (float*) alloc((size_t)BB*TT*DD*4);
  float*  q2c   = (float*) alloc((size_t)BB*DD*4);
  bf16_t* gbf   = (bf16_t*)alloc((size_t)BB*TT*1024*2);
  bf16_t* pre   = (bf16_t*)alloc((size_t)BB*TT*1024*2);   // reused for both layers
  bf16_t* mbf   = (bf16_t*)alloc((size_t)BB*TT*256*2);
  bf16_t* m2bf  = (bf16_t*)alloc((size_t)BB*TT*256*2);
  bf16_t* w1f   = (bf16_t*)alloc((size_t)512*1024*2);
  bf16_t* w1b   = (bf16_t*)alloc((size_t)512*1024*2);
  bf16_t* w2f   = (bf16_t*)alloc((size_t)512*256*2);
  bf16_t* w2b   = (bf16_t*)alloc((size_t)512*256*2);
  bf16_t* whh1f = (bf16_t*)alloc((size_t)512*128*2);
  bf16_t* whh1b = (bf16_t*)alloc((size_t)512*128*2);
  bf16_t* whh2f = (bf16_t*)alloc((size_t)512*128*2);
  bf16_t* whh2b = (bf16_t*)alloc((size_t)512*128*2);
  float*  bias1f = (float*)alloc(512*4);
  float*  bias1b = (float*)alloc(512*4);
  float*  bias2f = (float*)alloc(512*4);
  float*  bias2b = (float*)alloc(512*4);
  float*  z1    = (float*) alloc((size_t)BB*TT*4);
  float*  z2    = (float*) alloc((size_t)BB*TT*4);

  (void)hipFuncSetAttribute((const void*)k_lstm,
      hipFuncAttributeMaxDynamicSharedMemorySize, 176128);

  // prep
  k_prep_q<<<(BB*JJ*DD)/256, 256, 0, stream>>>(q, qbf, qTbf);
  k_prep_c<<<(BB*TT*DD)/256, 256, 0, stream>>>(c, wcq, cwbf);
  k_rowdot<<<(BB*TT+255)/256, 256, 0, stream>>>(c, wc, bc, bcq, cw, BB*TT);
  k_rowdot<<<(BB*JJ+255)/256, 256, 0, stream>>>(q, wq, bq, nullptr, qw, BB*JJ);
  k_f2bf<<<(512*1024+255)/256,256,0,stream>>>(l1f_Wih, w1f, 512*1024);
  k_f2bf<<<(512*1024+255)/256,256,0,stream>>>(l1b_Wih, w1b, 512*1024);
  k_f2bf<<<(512*256+255)/256,256,0,stream>>>(l2f_Wih, w2f, 512*256);
  k_f2bf<<<(512*256+255)/256,256,0,stream>>>(l2b_Wih, w2b, 512*256);
  k_f2bf<<<(512*128+255)/256,256,0,stream>>>(l1f_Whh, whh1f, 512*128);
  k_f2bf<<<(512*128+255)/256,256,0,stream>>>(l1b_Whh, whh1b, 512*128);
  k_f2bf<<<(512*128+255)/256,256,0,stream>>>(l2f_Whh, whh2f, 512*128);
  k_f2bf<<<(512*128+255)/256,256,0,stream>>>(l2b_Whh, whh2b, 512*128);
  k_addvec<<<2,256,0,stream>>>(l1f_bih, l1f_bhh, bias1f, 512);
  k_addvec<<<2,256,0,stream>>>(l1b_bih, l1b_bhh, bias1b, 512);
  k_addvec<<<2,256,0,stream>>>(l2f_bih, l2f_bhh, bias2f, 512);
  k_addvec<<<2,256,0,stream>>>(l2b_bih, l2b_bhh, bias2b, 512);

  // attention flow
  k_attn<<<(BB*TT/16)/4, 128, 0, stream>>>(cwbf, qbf, qTbf, cw, qw, smax, c2q);
  k_q2c<<<BB, 256, 0, stream>>>(smax, c, q2c);
  k_g<<<(BB*TT*1024)/256, 256, 0, stream>>>(c, c2q, q2c, gbf);

  // layer 1: input projection GEMMs + recurrence
  k_gemm<<<dim3(BB*TT/64, 4), 256, 0, stream>>>(gbf, 1024, w1f, bias1f, pre, 0);
  k_gemm<<<dim3(BB*TT/64, 4), 256, 0, stream>>>(gbf, 1024, w1b, bias1b, pre, 512);
  k_lstm<<<8, 256, 176128, stream>>>(pre, whh1f, whh1b, mbf);

  // layer 2
  k_gemm<<<dim3(BB*TT/64, 4), 256, 0, stream>>>(mbf, 256, w2f, bias2f, pre, 0);
  k_gemm<<<dim3(BB*TT/64, 4), 256, 0, stream>>>(mbf, 256, w2b, bias2b, pre, 512);
  k_lstm<<<8, 256, 176128, stream>>>(pre, whh2f, whh2b, m2bf);

  // pointer distributions
  k_dot<<<(BB*TT)/256, 256, 0, stream>>>(gbf, mbf, m2bf, W0, b0, W1, b1, z1, z2);
  k_smaxT<<<2*BB, 256, 0, stream>>>(z1, z2, (float*)d_out);
}
